// PaddedFFTConvInference_24936580121251
// MI455X (gfx1250) — matrix operand order repaired and checked by reading, not compiled, not run on hardware
//
#include <hip/hip_runtime.h>
#include <hip/hip_bf16.h>

// ---------------------------------------------------------------------------
// PaddedFFTConvInference on MI455X (gfx1250): chunked state-space formulation.
//   K[n,l] = sum_s E[n,s] * Re(z^l),  z = exp(dtA_real + i*dtA_imag)
//   intra-chunk (Q=64) : lower-tri Toeplitz of K[0..63]        (bf16 WMMA)
//   inter-chunk        : H[c+1] = z^Q H[c] + W x_chunk         (fp32 scan)
//                        y += Ure*ReH + Uim*ImH                (bf16 WMMA)
// conv_kernel: A tiles via TENSOR_LOAD_TO_LDS (TDM, padded deposit),
//              B tiles via GLOBAL_LOAD_ASYNC_TO_LDS_B128, both double-buffered
//              and pipelined against the WMMAs (asynccnt/tensorcnt waits).
// proj_gemm:   double-buffered LDS, b128 global loads, b64 ds stores.
// ---------------------------------------------------------------------------

typedef __bf16 bf16;
typedef __attribute__((ext_vector_type(16))) __bf16 v16bf;
typedef __attribute__((ext_vector_type(8)))  __bf16 v8bf;
typedef __attribute__((ext_vector_type(4)))  __bf16 v4bf;
typedef __attribute__((ext_vector_type(8)))  float  v8f;
typedef __attribute__((ext_vector_type(4)))  unsigned int v4u;
typedef __attribute__((ext_vector_type(8)))  int v8i;
typedef __attribute__((ext_vector_type(4)))  int v4i;

namespace {
constexpr int BSZ = 4, CDIM = 512, NDIM = 512, SDIM = 64, LEN = 8192;
constexpr int Q = 64, NCHUNK = LEN / Q;      // 128 chunks
constexpr int LDA  = 40;                     // LDS pitch, 80B (16B aligned)
constexpr int LDAT = 72;                     // TDM-padded pitch, 144B
}

__device__ inline bf16 to_bf(float v) { return (bf16)v; }
__device__ inline bf16 to_bf(bf16 v)  { return v; }
__device__ inline void store_out(float* p, float v) { __builtin_nontemporal_store(v, p); }
__device__ inline void store_out(bf16*  p, float v) { *p = (bf16)v; }

// CDNA5 async copy: LDS[lds_off] = MEM[gaddr], 16 bytes, tracked by ASYNCcnt.
__device__ inline void async_ldst_b128(unsigned lds_off, const void* gaddr) {
  asm volatile("global_load_async_to_lds_b128 %0, %1, off"
               :: "v"(lds_off), "v"((unsigned long long)gaddr) : "memory");
}

// TDM: load a 64x64 bf16 tile (row stride 64) into LDS with 16B padding per
// 128B row -> deposited pitch 144B (LDAT). Tracked by TENSORcnt.
__device__ inline void tdm_load_tile64(unsigned lds_addr, const bf16* gptr) {
  unsigned long long ga = (unsigned long long)gptr;
  v4u g0 = { 1u,                                   // count=1, gather off
             lds_addr,                             // D#.lds_addr
             (unsigned)ga,                         // global_addr[31:0]
             (unsigned)((ga >> 32) & 0x01FFFFFFu) | (2u << 30) }; // +type=2
  v8i g1 = { (int)((1u << 16) |                    // data_size = 2B
                   (1u << 20) |                    // pad_enable
                   (4u << 22) |                    // pad_interval: 32 DW=128B
                   (3u << 25)),                    // pad_amount: 4 DW=16B
             (int)(64u << 16),                     // tensor_dim0 = 64
             (int)(64u << 16),                     // tensor_dim1 = 64 (lo16)
             (int)(64u << 16),                     // tile_dim0 = 64
             64,                                   // tile_dim1 = 64
             64,                                   // tensor_dim0_stride = 64
             (int)(4096u << 16),                   // tensor_dim1_stride lo16
             0 };
  v4i gz = { 0, 0, 0, 0 };
  v8i gz8 = { 0, 0, 0, 0, 0, 0, 0, 0 };
  __builtin_amdgcn_tensor_load_to_lds(g0, g1, gz, gz, gz8, 0);
}

// A-fragment (16x32 bf16, M x K), pitch LD elements (LD*2 % 16 == 0).
template <int LD>
__device__ inline v16bf frag_a(const bf16* tile) {
  static_assert((LD * 2) % 16 == 0, "pitch must be 16B aligned");
  const int lane = threadIdx.x & 31;
  const bf16* p = tile + (lane & 15) * LD + ((lane >> 4) << 3);
  v8bf lo = *(const v8bf*)p;
  v8bf hi = *(const v8bf*)(p + 16);
  return __builtin_shufflevector(lo, hi, 0,1,2,3,4,5,6,7,8,9,10,11,12,13,14,15);
}

// B-fragment (32x16 bf16, K x N) from a K-contiguous LDS tile (pitch LDA).
__device__ inline v16bf frag_b(const bf16* tileT) {
  const int lane = threadIdx.x & 31;
  const bf16* p = tileT + (lane & 15) * LDA + ((lane >> 4) << 3);
  v8bf lo = *(const v8bf*)p;
  v8bf hi = *(const v8bf*)(p + 16);
  return __builtin_shufflevector(lo, hi, 0,1,2,3,4,5,6,7,8,9,10,11,12,13,14,15);
}

// --------------------------------------------------------------------------
// Kernel 1: per-(n,s) tables via complex power recurrence.
// --------------------------------------------------------------------------
__global__ __launch_bounds__(256)
void precompute_ns(const float* __restrict__ ar_, const float* __restrict__ ai_,
                   const float* __restrict__ E_, float* __restrict__ Wre,
                   float* __restrict__ Wim, float* __restrict__ zQr,
                   float* __restrict__ zQi, bf16* __restrict__ Acat) {
  int idx = blockIdx.x * blockDim.x + threadIdx.x;   // n*SDIM + s
  if (idx >= NDIM * SDIM) return;
  int n = idx >> 6, s = idx & 63;
  float ar = ar_[idx], ai = ai_[idx], e = E_[idx];
  float er = __expf(ar), c0 = __cosf(ai), s0 = __sinf(ai);
  float pr = 1.f, pi = 0.f;                           // z^p
  for (int p = 0; p <= Q; ++p) {
    if (p < Q) {
      Wre[(size_t)idx * Q + (Q - 1 - p)] = pr;        // z^{Q-1-j}
      Wim[(size_t)idx * Q + (Q - 1 - p)] = pi;
    }
    if (p >= 1) {
      int i = p - 1;
      Acat[((size_t)n * 3 + 1) * Q * Q + i * Q + s] = (bf16)( e * pr);
      Acat[((size_t)n * 3 + 2) * Q * Q + i * Q + s] = (bf16)(-e * pi);
    }
    if (p == Q) { zQr[idx] = pr; zQi[idx] = pi; }
    float nr = er * (pr * c0 - pi * s0);
    float ni = er * (pr * s0 + pi * c0);
    pr = nr; pi = ni;
  }
}

// Kernel 2: K64[n][p] = sum_s E * exp(ar*p) * cos(ai*p), p < Q.
__global__ __launch_bounds__(256)
void compute_k64(const float* __restrict__ ar_, const float* __restrict__ ai_,
                 const float* __restrict__ E_, float* __restrict__ K64) {
  int idx = blockIdx.x * blockDim.x + threadIdx.x;
  if (idx >= NDIM * Q) return;
  int n = idx >> 6;
  float p = (float)(idx & 63);
  float acc = 0.f;
  for (int s = 0; s < SDIM; ++s) {
    int k = n * SDIM + s;
    acc += E_[k] * __expf(ar_[k] * p) * __cosf(ai_[k] * p);
  }
  K64[idx] = acc;
}

// Kernel 3: lower-triangular Toeplitz block Acat[n][0][i][j] = K64[n][i-j].
__global__ __launch_bounds__(256)
void build_toeplitz(const float* __restrict__ K64, bf16* __restrict__ Acat) {
  int idx = blockIdx.x * blockDim.x + threadIdx.x;
  if (idx >= NDIM * Q * Q) return;
  int n = idx >> 12, i = (idx >> 6) & 63, j = idx & 63;
  Acat[((size_t)n * 3) * Q * Q + i * Q + j] =
      (bf16)((i >= j) ? K64[n * Q + (i - j)] : 0.f);
}

// --------------------------------------------------------------------------
// Kernel 4/7: Out[b][m][l] = sum_k A[m][k] * Bsrc[b][k][l].
// 128x128 tile, 8 waves (2x4), double-buffered LDS, vectorized staging.
// --------------------------------------------------------------------------
template <typename TB, typename TOUT>
__global__ __launch_bounds__(256)
void proj_gemm(const float* __restrict__ A, const TB* __restrict__ Bsrc,
               TOUT* __restrict__ Out) {
  constexpr int K = 512, M = 512, N = LEN;
  __shared__ bf16 As[2][128][LDA];
  __shared__ bf16 BsT[2][128][LDA];          // K-contiguous per column
  const int tid = threadIdx.x, lane = tid & 31, wid = tid >> 5;
  const int wm = wid & 1, wn = wid >> 1;
  const int m0 = blockIdx.y * 128, n0 = blockIdx.x * 128;
  const size_t boff = (size_t)blockIdx.z * K * N;
  const size_t ooff = (size_t)blockIdx.z * M * N;

  auto stage = [&](int buf, int kk) {
    #pragma unroll
    for (int t = 0; t < 4; ++t) {            // A tile 128x32: float4 -> 4xbf16
      int e = tid + t * 256;
      int row = e >> 3, c4 = (e & 7) * 4;
      float4 v = *(const float4*)&A[(size_t)(m0 + row) * K + kk + c4];
      v4bf w = { (bf16)v.x, (bf16)v.y, (bf16)v.z, (bf16)v.w };
      *(v4bf*)&As[buf][row][c4] = w;
    }
    #pragma unroll
    for (int t = 0; t < 4; ++t) {            // B tile 32x128 -> K-contig LDS
      int e = tid + t * 256;
      int col = e & 127, kg = (e >> 7) * 4;
      const TB* bp = &Bsrc[boff + (size_t)(kk + kg) * N + n0 + col];
      v4bf w = { to_bf(bp[0]), to_bf(bp[(size_t)N]),
                 to_bf(bp[2 * (size_t)N]), to_bf(bp[3 * (size_t)N]) };
      *(v4bf*)&BsT[buf][col][kg] = w;
    }
  };

  v8f acc[4][2];
  v8f vz = {};
  #pragma unroll
  for (int i = 0; i < 4; ++i)
    #pragma unroll
    for (int j = 0; j < 2; ++j) acc[i][j] = vz;

  stage(0, 0);
  __syncthreads();
  int buf = 0;
  for (int kk = 0; kk < K; kk += 32) {
    if (kk + 32 < K) stage(buf ^ 1, kk + 32);
    v16bf bfrag[2];
    #pragma unroll
    for (int j = 0; j < 2; ++j) bfrag[j] = frag_b(&BsT[buf][wn * 32 + j * 16][0]);
    #pragma unroll
    for (int i = 0; i < 4; ++i) {
      v16bf af = frag_a<LDA>(&As[buf][wm * 64 + i * 16][0]);
      #pragma unroll
      for (int j = 0; j < 2; ++j)
        acc[i][j] = __builtin_amdgcn_wmma_f32_16x16x32_bf16(
            false, af, false, bfrag[j], (short)0, acc[i][j], false, false);
    }
    __syncthreads();
    buf ^= 1;
  }
  const int colw = lane & 15, mo = (lane >> 4) * 8;
  #pragma unroll
  for (int i = 0; i < 4; ++i)
    #pragma unroll
    for (int j = 0; j < 2; ++j) {
      int row = m0 + wm * 64 + i * 16 + mo;
      int col = n0 + wn * 32 + j * 16 + colw;
      #pragma unroll
      for (int r = 0; r < 8; ++r)
        store_out(&Out[ooff + (size_t)(row + r) * N + col], acc[i][j][r]);
    }
}

// --------------------------------------------------------------------------
// Kernel 5: state scan. One block per n, thread = (b, s). W cached in LDS.
// --------------------------------------------------------------------------
__global__ __launch_bounds__(256)
void scan_kernel(const float* __restrict__ Wre, const float* __restrict__ Wim,
                 const float* __restrict__ zQr_, const float* __restrict__ zQi_,
                 const bf16* __restrict__ xb, bf16* __restrict__ Hre,
                 bf16* __restrict__ Him) {
  __shared__ float Wr_s[SDIM][Q + 1];
  __shared__ float Wi_s[SDIM][Q + 1];
  __shared__ float xsh[BSZ][Q];
  const int n = blockIdx.x, tid = threadIdx.x;
  const int b = tid >> 6, s = tid & 63;
  for (int e = tid; e < SDIM * Q; e += 256) {
    int ss = e >> 6, j = e & 63;
    Wr_s[ss][j] = Wre[((size_t)n * SDIM + ss) * Q + j];
    Wi_s[ss][j] = Wim[((size_t)n * SDIM + ss) * Q + j];
  }
  __syncthreads();
  const float zr = zQr_[n * SDIM + s], zi = zQi_[n * SDIM + s];
  float hr = 0.f, hi = 0.f;
  const bf16* xp = xb + ((size_t)b * NDIM + n) * LEN;
  for (int c = 0; c < NCHUNK; ++c) {
    xsh[b][s] = (float)xp[c * Q + s];
    __syncthreads();
    size_t ho = (((size_t)b * NDIM + n) * NCHUNK + c) * SDIM + s;
    Hre[ho] = (bf16)hr;                   // state *entering* chunk c
    Him[ho] = (bf16)hi;
    float arr = 0.f, aii = 0.f;
    #pragma unroll 8
    for (int j = 0; j < Q; ++j) {
      float xv = xsh[b][j];
      arr += Wr_s[s][j] * xv;
      aii += Wi_s[s][j] * xv;
    }
    float nhr = zr * hr - zi * hi + arr;
    float nhi = zr * hi + zi * hr + aii;
    hr = nhr; hi = nhi;
    __syncthreads();
  }
}

// --------------------------------------------------------------------------
// Kernel 6: chunked conv. Per n: Y[64 x 512] = T*X + Ure*ReH + Uim*ImH.
// 6-step fully unrolled pipeline, A tiles via TDM, B tiles via async-to-LDS,
// both double-buffered; compute overlaps the in-flight copies.
// --------------------------------------------------------------------------
__global__ __launch_bounds__(256)
void conv_kernel(const bf16* __restrict__ Acat, const bf16* __restrict__ xb,
                 const bf16* __restrict__ Hre, const bf16* __restrict__ Him,
                 bf16* __restrict__ yb) {
  __shared__ bf16 Abuf[2][Q][LDAT];          // TDM-deposited (padded pitch)
  __shared__ bf16 Bbuf[2][128][LDA];
  const int n = blockIdx.y;
  const int n0 = blockIdx.x * 128;
  const int tid = threadIdx.x, lane = tid & 31, wid = tid >> 5;
  const int wm = wid & 3, wn = wid >> 2;     // 4 M-waves x 2 N-waves
  const unsigned ldsA[2] = { (unsigned)(size_t)&Abuf[0][0][0],
                             (unsigned)(size_t)&Abuf[1][0][0] };
  const unsigned ldsB[2] = { (unsigned)(size_t)&Bbuf[0][0][0],
                             (unsigned)(size_t)&Bbuf[1][0][0] };

  auto issue_B = [&](int step, unsigned dst) {   // 2 async instr per wave
    const int kb = step >> 1, ks = step & 1;
    #pragma unroll
    for (int t = 0; t < 2; ++t) {
      int e = tid + t * 256;
      int col = e >> 2, kg = (e & 3) * 8;
      int colg = n0 + col, b = colg >> 7, c = colg & 127;
      int kk = ks * 32 + kg;
      const bf16* g;
      if (kb == 0)      g = &xb[((size_t)b * NDIM + n) * LEN + c * Q + kk];
      else if (kb == 1) g = &Hre[(((size_t)b * NDIM + n) * NCHUNK + c) * SDIM + kk];
      else              g = &Him[(((size_t)b * NDIM + n) * NCHUNK + c) * SDIM + kk];
      async_ldst_b128(dst + (unsigned)(col * LDA + kg) * 2, g);
    }
  };
  auto issue_A = [&](int kb, unsigned dst) {     // one TDM op, wave 0 only
    if (wid == 0) tdm_load_tile64(dst, Acat + ((size_t)n * 3 + kb) * Q * Q);
  };

  v8f acc[4];
  v8f vz = {};
  #pragma unroll
  for (int j = 0; j < 4; ++j) acc[j] = vz;

  issue_A(0, ldsA[0]);
  issue_B(0, ldsB[0]);
  #pragma unroll
  for (int s = 0; s < 6; ++s) {
    const int kb = s >> 1, ks = s & 1;
    const int pa = kb & 1, pb = s & 1;
    if (s > 0) __syncthreads();                // prior reads done before reuse
    if (s < 5) issue_B(s + 1, ldsB[pb ^ 1]);
    if (ks == 1 && kb < 2) issue_A(kb + 1, ldsA[pa ^ 1]);
    if (s < 5) asm volatile("s_wait_asynccnt 0x2" ::: "memory");
    else       asm volatile("s_wait_asynccnt 0x0" ::: "memory");
    if (ks == 1 && kb < 2) {                   // one A-tile TDM still in flight
      if (wid == 0) __builtin_amdgcn_s_wait_tensorcnt(1);
    } else {
      if (wid == 0) __builtin_amdgcn_s_wait_tensorcnt(0);
    }
    __syncthreads();                           // tile s visible to all waves
    v16bf af = frag_a<LDAT>(&Abuf[pa][wm * 16][ks * 32]);
    #pragma unroll
    for (int j = 0; j < 4; ++j) {
      v16bf bfv = frag_b(&Bbuf[pb][wn * 64 + j * 16][0]);
      acc[j] = __builtin_amdgcn_wmma_f32_16x16x32_bf16(
          false, af, false, bfv, (short)0, acc[j], false, false);
    }
  }
  const int colw = lane & 15, mo = (lane >> 4) * 8;
  #pragma unroll
  for (int j = 0; j < 4; ++j) {                // 8 rows contiguous in y
    int colg = n0 + wn * 64 + j * 16 + colw;
    int b = colg >> 7, c = colg & 127;
    v8bf w;
    #pragma unroll
    for (int r = 0; r < 8; ++r) w[r] = (bf16)acc[j][r];
    *(v8bf*)&yb[((size_t)b * NDIM + n) * LEN + c * Q + wm * 16 + mo] = w;
  }
}

// --------------------------------------------------------------------------
extern "C" void kernel_launch(void* const* d_in, const int* in_sizes, int n_in,
                              void* d_out, int out_size, void* d_ws, size_t ws_size,
                              hipStream_t stream) {
  (void)in_sizes; (void)n_in; (void)out_size; (void)ws_size;
  const float* inp  = (const float*)d_in[0];   // [4,512,8192]
  const float* dtAr = (const float*)d_in[1];   // [512,64]
  const float* dtAi = (const float*)d_in[2];   // [512,64]
  const float* Bm   = (const float*)d_in[3];   // [512,512]
  const float* Cm   = (const float*)d_in[4];   // [512,512]
  const float* Em   = (const float*)d_in[5];   // [512,64]
  float* out = (float*)d_out;                  // [4,512,8192]

  char* ws = (char*)d_ws;
  auto take = [&](size_t bytes) {
    char* p = ws; ws += (bytes + 255) & ~(size_t)255; return p;
  };
  bf16*  xb   = (bf16*) take((size_t)BSZ * NDIM * LEN * 2);
  bf16*  yb   = (bf16*) take((size_t)BSZ * NDIM * LEN * 2);
  bf16*  Hre  = (bf16*) take((size_t)BSZ * NDIM * NCHUNK * SDIM * 2);
  bf16*  Him  = (bf16*) take((size_t)BSZ * NDIM * NCHUNK * SDIM * 2);
  float* Wre  = (float*)take((size_t)NDIM * SDIM * Q * 4);
  float* Wim  = (float*)take((size_t)NDIM * SDIM * Q * 4);
  float* zQr  = (float*)take((size_t)NDIM * SDIM * 4);
  float* zQi  = (float*)take((size_t)NDIM * SDIM * 4);
  float* K64  = (float*)take((size_t)NDIM * Q * 4);
  bf16*  Acat = (bf16*) take((size_t)NDIM * 3 * Q * Q * 2);

  precompute_ns <<<NDIM * SDIM / 256, 256, 0, stream>>>(dtAr, dtAi, Em, Wre, Wim, zQr, zQi, Acat);
  compute_k64   <<<NDIM * Q / 256,    256, 0, stream>>>(dtAr, dtAi, Em, K64);
  build_toeplitz<<<NDIM * Q * Q / 256,256, 0, stream>>>(K64, Acat);
  proj_gemm<float, bf16><<<dim3(LEN / 128, NDIM / 128, BSZ), 256, 0, stream>>>(Bm, inp, xb);
  scan_kernel   <<<NDIM, 256, 0, stream>>>(Wre, Wim, zQr, zQi, xb, Hre, Him);
  conv_kernel   <<<dim3((BSZ * NCHUNK) / 128, NDIM), 256, 0, stream>>>(Acat, xb, Hre, Him, yb);
  proj_gemm<bf16, float><<<dim3(LEN / 128, CDIM / 128, BSZ), 256, 0, stream>>>(Cm, yb, out);
}
